// MinibatchDiscrimination_55044300866237
// MI455X (gfx1250) — compile-verified, matrix operand order checked by reading, not run on hardware
//
#include <hip/hip_runtime.h>
#include <hip/hip_bf16.h>

typedef __attribute__((ext_vector_type(16))) _Float16 v16h;
typedef __attribute__((ext_vector_type(8)))  float    v8f;

#define B_SZ     256
#define IN_FEAT  256
#define OUT_FEAT 64
#define KERN     32
#define NCOL     (OUT_FEAT * KERN)   // 2048

// ---------------------------------------------------------------------------
// 1) BatchNorm statistics: per-feature scale/shift (training-mode, biased var)
// ---------------------------------------------------------------------------
__global__ void bn_stats(const float* __restrict__ x, const float* __restrict__ gamma,
                         const float* __restrict__ beta, float* __restrict__ scl,
                         float* __restrict__ shf) {
    const int f = threadIdx.x;                     // one feature per thread (256)
    float s = 0.f, s2 = 0.f;
    for (int b = 0; b < B_SZ; ++b) {
        float v = x[b * IN_FEAT + f];              // coalesced across threads
        s += v; s2 += v * v;
    }
    float mu  = s  * (1.f / B_SZ);
    float var = s2 * (1.f / B_SZ) - mu * mu;
    float sc  = gamma[f] * rsqrtf(var + 1e-5f);
    scl[f] = sc;
    shf[f] = beta[f] - mu * sc;
}

// ---------------------------------------------------------------------------
// 2) Normalize x and convert to f16 (WMMA A operand source)
// ---------------------------------------------------------------------------
__global__ void pack_x(const float* __restrict__ x, const float* __restrict__ scl,
                       const float* __restrict__ shf, _Float16* __restrict__ xh) {
    int i = blockIdx.x * 256 + threadIdx.x;        // 65536 elements
    int f = i & (IN_FEAT - 1);
    xh[i] = (_Float16)(x[i] * scl[f] + shf[f]);
}

// ---------------------------------------------------------------------------
// 3) Transpose T [K=256, N=2048] -> Tt [N=2048, K=256] in f16 so B fragments
//    load as contiguous 16-byte runs per lane.
// ---------------------------------------------------------------------------
__global__ void pack_T(const float* __restrict__ T, _Float16* __restrict__ Tt) {
    int i = blockIdx.x * 256 + threadIdx.x;        // 524288, i = k*2048 + n (coalesced read)
    int k = i >> 11, n = i & (NCOL - 1);
    Tt[n * IN_FEAT + k] = (_Float16)T[i];
}

// ---------------------------------------------------------------------------
// 4) GEMM via v_wmma_f32_16x16x32_f16.
//    One wave per 16x16 output tile; 8 waves/block; 256 blocks = 2048 tiles.
//    A/B lane layout (16-bit operands, wave32):
//      lanes 0-15  -> rows m0..m0+15 (A) / cols n0..n0+15 (B), K = base+0..7 and base+16..23
//      lanes 16-31 -> same rows/cols,                        K = base+8..15 and base+24..31
//    i.e. two contiguous 8-half (16B) runs per lane per K-step of 32.
// ---------------------------------------------------------------------------
__global__ void gemm_wmma(const _Float16* __restrict__ xh,   // [256, 256]
                          const _Float16* __restrict__ Tt,   // [2048, 256]
                          float* __restrict__ Mf)            // [256, 2048]
{
    const int tid  = threadIdx.x;
    const int wave = tid >> 5;
    const int lane = tid & 31;
    const int tile = blockIdx.x * 8 + wave;        // 0..2047
    const int m0 = (tile >> 7) * 16;               // batch-row tile
    const int n0 = (tile & 127) * 16;              // column tile

    const int lhalf = lane >> 4;                   // 0: K 0-7,16-23; 1: K 8-15,24-31
    const int l16   = lane & 15;
    const _Float16* arow = xh + (m0 + l16) * IN_FEAT;
    const _Float16* brow = Tt + (n0 + l16) * IN_FEAT;

    v8f c = {};
#pragma unroll
    for (int kk = 0; kk < IN_FEAT / 32; ++kk) {
        const int kb = kk * 32 + lhalf * 8;
        v16h a, b;
        *((uint4*)&a)       = *(const uint4*)(arow + kb);        // K = kb..kb+7
        *(((uint4*)&a) + 1) = *(const uint4*)(arow + kb + 16);   // K = kb+16..kb+23
        *((uint4*)&b)       = *(const uint4*)(brow + kb);
        *(((uint4*)&b) + 1) = *(const uint4*)(brow + kb + 16);
        c = __builtin_amdgcn_wmma_f32_16x16x32_f16(
                /*neg_a=*/false, a, /*neg_b=*/false, b,
                /*c_mod=*/(short)0, c, /*reuse_a=*/false, /*reuse_b=*/false);
    }

    // D layout: VGPR r -> M = m0 + r + 8*lhalf, N = n0 + l16
    const int orow = m0 + lhalf * 8;
    const int ocol = n0 + l16;
#pragma unroll
    for (int r = 0; r < 8; ++r)
        Mf[(orow + r) * NCOL + ocol] = c[r];
}

// ---------------------------------------------------------------------------
// 5) Pairwise L1 over the kernel dim. One block per output channel o.
//    M[:,o,:] (256x32 f32 = 32KB) staged in LDS; each thread b keeps its own
//    row in registers and streams all 256 rows (LDS broadcast reads).
// ---------------------------------------------------------------------------
__global__ void pairwise_l1(const float* __restrict__ Mf,   // [256, 2048]
                            float* __restrict__ ob)         // [64, 256]
{
    __shared__ float sm[B_SZ * KERN];              // 32 KB
    const int o   = blockIdx.x;                    // 0..63
    const int tid = threadIdx.x;                   // 0..255

    for (int i = tid; i < B_SZ * KERN; i += 256) { // each wave: 128B contiguous line
        int b = i >> 5, k = i & 31;
        sm[i] = Mf[b * NCOL + o * KERN + k];
    }
    __syncthreads();

    const float4* sm4 = (const float4*)sm;
    float4 r[8];
#pragma unroll
    for (int j = 0; j < 8; ++j) r[j] = sm4[tid * 8 + j];

    float acc = 0.f;
    for (int a = 0; a < B_SZ; ++a) {
#pragma unroll
        for (int j = 0; j < 8; ++j) {
            float4 v = sm4[a * 8 + j];             // broadcast: all lanes same addr
            acc += fabsf(r[j].x - v.x) + fabsf(r[j].y - v.y) +
                   fabsf(r[j].z - v.z) + fabsf(r[j].w - v.w);
        }
    }
    ob[o * B_SZ + tid] = acc;
}

// ---------------------------------------------------------------------------
// 6) Final EqualLinear: out[b] = (sum_o ob[o,b] * W1[o]) * sqrt(2/64) + b1
// ---------------------------------------------------------------------------
__global__ void finalize(const float* __restrict__ ob, const float* __restrict__ W1,
                         const float* __restrict__ b1, float* __restrict__ out) {
    const int b = threadIdx.x;
    float acc = 0.f;
    for (int o = 0; o < OUT_FEAT; ++o)
        acc += ob[o * B_SZ + b] * W1[o];           // coalesced across threads
    out[b] = acc * 0.17677669529663687f /* sqrt(2/64) */ + b1[0];
}

// ---------------------------------------------------------------------------
extern "C" void kernel_launch(void* const* d_in, const int* in_sizes, int n_in,
                              void* d_out, int out_size, void* d_ws, size_t ws_size,
                              hipStream_t stream) {
    const float* feature = (const float*)d_in[0];  // [256, 256]
    const float* T       = (const float*)d_in[1];  // [256, 64, 32]
    const float* gamma   = (const float*)d_in[2];  // [256]
    const float* beta    = (const float*)d_in[3];  // [256]
    const float* W1      = (const float*)d_in[4];  // [1, 64]
    const float* b1      = (const float*)d_in[5];  // [1]
    float* out = (float*)d_out;                    // [256, 1]

    char* w = (char*)d_ws;
    float*     scl = (float*)(w + 0);                    // 1 KB
    float*     shf = (float*)(w + 1024);                 // 1 KB
    _Float16*  xh  = (_Float16*)(w + 4096);              // 128 KB
    _Float16*  Tt  = (_Float16*)(w + (256u << 10));      // 1 MB  @ 262144
    float*     Mf  = (float*)(w + (1280u << 10));        // 2 MB  @ 1310720
    float*     ob  = (float*)(w + (1280u << 10) + (2048u << 10)); // 64 KB @ 3407872

    bn_stats   <<<1,    256, 0, stream>>>(feature, gamma, beta, scl, shf);
    pack_x     <<<256,  256, 0, stream>>>(feature, scl, shf, xh);
    pack_T     <<<2048, 256, 0, stream>>>(T, Tt);
    gemm_wmma  <<<256,  256, 0, stream>>>(xh, Tt, Mf);
    pairwise_l1<<<64,   256, 0, stream>>>(Mf, ob);
    finalize   <<<1,    256, 0, stream>>>(ob, W1, b1, out);
}
